// Net_8907762172415
// MI455X (gfx1250) — matrix operand order repaired and checked by reading, not compiled, not run on hardware
//
#include <hip/hip_runtime.h>

typedef __attribute__((ext_vector_type(16))) _Float16 v16h;
typedef __attribute__((ext_vector_type(8)))  _Float16 v8h;
typedef __attribute__((ext_vector_type(8)))  float    v8f;

#define BN_INV 0.9999950000374997f

__device__ __forceinline__ float eluf(float v) { return v > 0.0f ? v : (expf(v) - 1.0f); }

// ---------------------------------------------------------------------------
// kNN: one thread per query point; maintain sorted top-(K*DIL) ascending by
// distance, ties -> lower index (matches jax top_k on -d2). Emit every DIL-th.
// ---------------------------------------------------------------------------
template<int K, int DIL>
__global__ void knn_kernel(const float* __restrict__ pos, int B, int P,
                           int* __restrict__ idx_out) {
  const int KK = K * DIL;
  int q = blockIdx.x * blockDim.x + threadIdx.x;
  if (q >= B * P) return;
  int b = q / P, p = q % P;
  const float* pb = pos + (size_t)b * P * 3;
  float qx = pb[p * 3 + 0], qy = pb[p * 3 + 1], qz = pb[p * 3 + 2];
  float bestd[KK];
  int   besti[KK];
#pragma unroll
  for (int i = 0; i < KK; i++) { bestd[i] = 3.4e38f; besti[i] = 0; }
  for (int j = 0; j < P; j++) {
    float dx = pb[j * 3 + 0] - qx;
    float dy = pb[j * 3 + 1] - qy;
    float dz = pb[j * 3 + 2] - qz;
    float d = dx * dx + dy * dy + dz * dz;
    if (d < bestd[KK - 1]) {
      int i = KK - 1;
      while (i > 0 && bestd[i - 1] > d) {
        bestd[i] = bestd[i - 1]; besti[i] = besti[i - 1]; --i;
      }
      bestd[i] = d; besti[i] = j;
    }
  }
  for (int j = 0; j < K; j++) idx_out[(size_t)q * K + j] = besti[j * DIL];
}

// ---------------------------------------------------------------------------
// Farthest point sampling: one block (256 thr) per batch. Deterministic,
// argmax ties -> lowest index (matches jnp.argmax).
// ---------------------------------------------------------------------------
__global__ void fps_kernel(const float* __restrict__ pos, int P, int n,
                           int* __restrict__ idx_out) {
  __shared__ float mind[2048];
  __shared__ float redv[256];
  __shared__ int   redi[256];
  __shared__ float curp[3];
  int b = blockIdx.x, t = threadIdx.x;
  const float* pb = pos + (size_t)b * P * 3;
  if (t == 0) {
    idx_out[(size_t)b * n + 0] = 0;
    curp[0] = pb[0]; curp[1] = pb[1]; curp[2] = pb[2];
  }
  __syncthreads();
  for (int p = t; p < P; p += 256) {
    float dx = pb[p * 3 + 0] - curp[0];
    float dy = pb[p * 3 + 1] - curp[1];
    float dz = pb[p * 3 + 2] - curp[2];
    mind[p] = dx * dx + dy * dy + dz * dz;
  }
  __syncthreads();
  for (int i = 1; i < n; i++) {
    float bv = -1.0f; int bi = P;
    for (int p = t; p < P; p += 256) {
      float v = mind[p];
      if (v > bv) { bv = v; bi = p; }
    }
    redv[t] = bv; redi[t] = bi;
    __syncthreads();
    for (int s = 128; s > 0; s >>= 1) {
      if (t < s) {
        float ov = redv[t + s]; int oi = redi[t + s];
        if (ov > redv[t] || (ov == redv[t] && oi < redi[t])) { redv[t] = ov; redi[t] = oi; }
      }
      __syncthreads();
    }
    int sel = redi[0];
    if (t == 0) {
      idx_out[(size_t)b * n + i] = sel;
      curp[0] = pb[sel * 3 + 0]; curp[1] = pb[sel * 3 + 1]; curp[2] = pb[sel * 3 + 2];
    }
    __syncthreads();
    for (int p = t; p < P; p += 256) {
      float dx = pb[p * 3 + 0] - curp[0];
      float dy = pb[p * 3 + 1] - curp[1];
      float dz = pb[p * 3 + 2] - curp[2];
      float d = dx * dx + dy * dy + dz * dz;
      mind[p] = fminf(mind[p], d);
    }
    __syncthreads();
  }
}

// dst[b,i,c] = src[b, idx[b,i], c]
__global__ void gather_kernel(const float* __restrict__ src, const int* __restrict__ idx,
                              float* __restrict__ dst, int B, int n, int C, int Psrc) {
  long i = (long)blockIdx.x * blockDim.x + threadIdx.x;
  if (i >= (long)B * n * C) return;
  int c = (int)(i % C);
  long t = i / C;
  int j = (int)(t % n);
  int b = (int)(t / n);
  dst[i] = src[((size_t)b * Psrc + idx[(size_t)b * n + j]) * C + c];
}

// rel[b,p,k,:] = pos[b,idx,:] - pos[b,p,:]
__global__ void rel_kernel(const float* __restrict__ pos, const int* __restrict__ idx,
                           float* __restrict__ rel, int B, int P, int K) {
  long i = (long)blockIdx.x * blockDim.x + threadIdx.x;
  if (i >= (long)B * P * K) return;
  long t = i / K;
  int p = (int)(t % P);
  int b = (int)(t / P);
  int nb = idx[i];
  const float* pb = pos + (size_t)b * P * 3;
  rel[i * 3 + 0] = pb[nb * 3 + 0] - pb[p * 3 + 0];
  rel[i * 3 + 1] = pb[nb * 3 + 1] - pb[p * 3 + 1];
  rel[i * 3 + 2] = pb[nb * 3 + 2] - pb[p * 3 + 2];
}

// mlp1 layer 1: (rows x 3) @ (3 x cd) + b -> elu -> bn, store f16
__global__ void mlp1a_kernel(const float* __restrict__ rel, const float* __restrict__ w1,
                             const float* __restrict__ b1, const float* __restrict__ g,
                             const float* __restrict__ beta, _Float16* __restrict__ out,
                             long rows, int cd) {
  long i = (long)blockIdx.x * blockDim.x + threadIdx.x;
  if (i >= rows * cd) return;
  int c = (int)(i % cd);
  long r = i / cd;
  const float* rp = rel + r * 3;
  float v = rp[0] * w1[0 * cd + c] + rp[1] * w1[1 * cd + c] + rp[2] * w1[2 * cd + c] + b1[c];
  v = g[c] * eluf(v) * BN_INV + beta[c];
  out[i] = (_Float16)v;
}

// f32 (Kd x N) weights -> f16 TRANSPOSED (N x Kpad), zero padded k >= Kd.
// Transposed layout makes the WMMA B-fragment (one column, 16 consecutive k)
// a single contiguous 32-byte vector load.
__global__ void convwT_kernel(const float* __restrict__ w, _Float16* __restrict__ out,
                              int Kd, int Kpad, int N) {
  long i = (long)blockIdx.x * blockDim.x + threadIdx.x;
  if (i >= (long)N * Kpad) return;
  int k  = (int)(i % Kpad);
  int nn = (int)(i / Kpad);
  out[i] = (_Float16)((k < Kd) ? w[(size_t)k * N + nn] : 0.0f);
}

// ---------------------------------------------------------------------------
// WMMA GEMM: C(MxN) = act(A(MxKd) @ B(KdxN) + bias), f16 in / f32 acc.
// B supplied TRANSPOSED (N x Kd). EPI==0 -> f16 out, elu+bn epilogue;
// EPI==1 -> f32 out, relu. Each wave: 4 M-tiles x 1 N-tile -> one B-fragment
// load feeds 4 back-to-back v_wmma ops. A and B fragments both use contiguous k = 16*(lane/16) + s.
// ---------------------------------------------------------------------------
template<int EPI>
__global__ void gemm_wmma_kernel(const _Float16* __restrict__ A, const _Float16* __restrict__ BT,
                                 const float* __restrict__ bias, const float* __restrict__ bng,
                                 const float* __restrict__ bnb, void* __restrict__ Cout,
                                 int M, int N, int Kd) {
  int wave = threadIdx.x >> 5;
  int lane = threadIdx.x & 31;
  int ntile = blockIdx.y * 8 + wave;
  if (ntile * 16 >= N) return;
  int mt0 = blockIdx.x * 4;
  int n = ntile * 16 + (lane & 15);
  int kbA = (lane < 16) ? 0 : 16;  // A: lanes 16-31 hold K=16..31
  int kbB = (lane < 16) ? 0 : 16;  // B: lanes 16-31 hold K=16..31
  const _Float16* btrow = BT + (size_t)n * Kd;

  const _Float16* ar[4];
#pragma unroll
  for (int t = 0; t < 4; t++) {
    int rb = ((mt0 + t) * 16 < M) ? (mt0 + t) * 16 : 0;  // clamp inactive tiles
    ar[t] = A + (size_t)(rb + (lane & 15)) * Kd;
  }
  v8f acc[4] = {{}, {}, {}, {}};
  for (int kk = 0; kk < Kd; kk += 32) {
    v16h bfrag = *(const v16h*)(btrow + kk + kbB);
    union { v16h v; v8h h[2]; } a[4];
#pragma unroll
    for (int t = 0; t < 4; t++) {
      a[t].h[0] = *(const v8h*)(ar[t] + kk + kbA);
      a[t].h[1] = *(const v8h*)(ar[t] + kk + kbA + 8);
    }
#pragma unroll
    for (int t = 0; t < 4; t++) {
      acc[t] = __builtin_amdgcn_wmma_f32_16x16x32_f16(false, a[t].v, false, bfrag,
                                                      (short)0, acc[t], false, false);
    }
  }
#pragma unroll
  for (int t = 0; t < 4; t++) {
    int mt = mt0 + t;
    if (mt * 16 >= M) break;
#pragma unroll
    for (int j = 0; j < 8; j++) {
      int mm = mt * 16 + ((lane < 16) ? j : j + 8);
      float v = acc[t][j] + bias[n];
      if constexpr (EPI == 0) {
        v = bng[n] * eluf(v) * BN_INV + bnb[n];
        ((_Float16*)Cout)[(size_t)mm * N + n] = (_Float16)v;
      } else {
        v = fmaxf(v, 0.0f);
        ((float*)Cout)[(size_t)mm * N + n] = v;
      }
    }
  }
}

// ---------------------------------------------------------------------------
// Per-point K x K transform (mlp2 chain). One block per point, K*K workers.
// ---------------------------------------------------------------------------
template<int K>
__global__ void tmat_kernel(const float* __restrict__ rel,
                            const float* __restrict__ w1, const float* __restrict__ b1,
                            const float* __restrict__ g3, const float* __restrict__ b3,
                            const float* __restrict__ cw1, const float* __restrict__ cb1,
                            const float* __restrict__ g4, const float* __restrict__ b4,
                            const float* __restrict__ cw2, const float* __restrict__ cb2,
                            const float* __restrict__ g5, const float* __restrict__ b5,
                            float* __restrict__ T) {
  int pt = blockIdx.x;
  int t = threadIdx.x;
  __shared__ float relf[3 * K];
  __shared__ float tb[K * K];
  __shared__ float tb2[K * K];
  if (t < 3 * K) relf[t] = rel[(size_t)pt * 3 * K + t];
  __syncthreads();
  int k = t / K, mcol = t % K;
  if (t < K * K) {
    float v = b1[t];
    for (int l = 0; l < 3 * K; l++) v += relf[l] * w1[(size_t)l * (K * K) + t];
    v = eluf(v);
    tb[t] = g3[t] * v * BN_INV + b3[t];
  }
  __syncthreads();
  if (t < K * K) {
    float v = cb1[t];
#pragma unroll
    for (int l = 0; l < K; l++) v += tb[k * K + l] * cw1[(size_t)(k * K + mcol) * K + l];
    v = eluf(v);
    tb2[t] = g4[t] * v * BN_INV + b4[t];
  }
  __syncthreads();
  if (t < K * K) {
    float v = cb2[t];
#pragma unroll
    for (int l = 0; l < K; l++) v += tb2[k * K + l] * cw2[(size_t)(k * K + mcol) * K + l];
    T[(size_t)pt * K * K + t] = g5[t] * v * BN_INV + b5[t];
  }
}

// ---------------------------------------------------------------------------
// Fused X-transform + depthwise conv producing the lin-GEMM feature row (f16,
// zero-padded to Kpad). One block per point; threads stride over channels.
// Channel c < cd comes from mlp1 output, c >= cd gathers previous features.
// ---------------------------------------------------------------------------
template<int K>
__global__ void xtdw_kernel(const _Float16* __restrict__ hmlp, const float* __restrict__ xprev,
                            const int* __restrict__ idx, const float* __restrict__ T,
                            const float* __restrict__ wdw, const float* __restrict__ db,
                            _Float16* __restrict__ feat, int B, int P,
                            int cd, int cin, int dm, int Kpad) {
  int pt = blockIdx.x;
  int b = pt / P;
  int t = threadIdx.x;
  __shared__ float Ts[K * K];
  __shared__ int idxs[K];
  if (t < K * K) Ts[t] = T[(size_t)pt * K * K + t];
  if (t < K) idxs[t] = idx[(size_t)pt * K + t];
  __syncthreads();
  int C = cd + cin;
  for (int c = t; c < C; c += blockDim.x) {
    float hv[K];
    if (c < cd) {
#pragma unroll
      for (int kk = 0; kk < K; kk++) hv[kk] = (float)hmlp[((size_t)pt * K + kk) * cd + c];
    } else {
      int cc = c - cd;
#pragma unroll
      for (int kk = 0; kk < K; kk++) hv[kk] = xprev[((size_t)b * P + idxs[kk]) * cin + cc];
    }
    float xt[K];
#pragma unroll
    for (int mcol = 0; mcol < K; mcol++) {
      float s = 0.0f;
#pragma unroll
      for (int kk = 0; kk < K; kk++) s += hv[kk] * Ts[kk * K + mcol];
      xt[mcol] = s;
    }
    for (int d = 0; d < dm; d++) {
      float s = db[c * dm + d];
#pragma unroll
      for (int mcol = 0; mcol < K; mcol++) s += xt[mcol] * wdw[((size_t)c * dm + d) * K + mcol];
      feat[(size_t)pt * Kpad + c * dm + d] = (_Float16)s;
    }
  }
  for (int j = C * dm + t; j < Kpad; j += blockDim.x)
    feat[(size_t)pt * Kpad + j] = (_Float16)0.0f;
}

__global__ void meanpool_kernel(const float* __restrict__ h, float* __restrict__ out,
                                int B, int P, int C) {
  int i = blockIdx.x * blockDim.x + threadIdx.x;
  if (i >= B * C) return;
  int b = i / C, c = i % C;
  float s = 0.0f;
  for (int p = 0; p < P; p++) s += h[((size_t)b * P + p) * C + c];
  out[i] = s / (float)P;
}

// ---------------------------------------------------------------------------
// Host-side orchestration
// ---------------------------------------------------------------------------
struct Scratch {
  int* idx; float* rel; _Float16* a; _Float16* hmlp; float* T;
  _Float16* feat; _Float16* w2h; _Float16* linw;
};

template<int K, int DIL>
static void run_layer(hipStream_t s, void* const* d_in, int layer,
                      const float* poscur, int P, const float* xprev, int cin,
                      int cd, int dm, int cout, const Scratch& sc, float* hout) {
  const int Bn = 16;
  const int BP = Bn * P;
  const int C = cin + cd;
  const int Cdm = C * dm;
  const int Kpad = (Cdm + 31) & ~31;
  const long rows = (long)BP * K;
  auto prm = [&](int i) { return (const float*)d_in[2 + layer * 24 + i]; };

  knn_kernel<K, DIL><<<dim3((BP + 63) / 64), dim3(64), 0, s>>>(poscur, Bn, P, sc.idx);
  rel_kernel<<<dim3((unsigned)(((long)BP * K + 255) / 256)), dim3(256), 0, s>>>(
      poscur, sc.idx, sc.rel, Bn, P, K);
  mlp1a_kernel<<<dim3((unsigned)((rows * cd + 255) / 256)), dim3(256), 0, s>>>(
      sc.rel, prm(0), prm(1), prm(2), prm(3), sc.a, rows, cd);
  convwT_kernel<<<dim3((unsigned)(((long)cd * cd + 255) / 256)), dim3(256), 0, s>>>(
      prm(4), sc.w2h, cd, cd, cd);
  gemm_wmma_kernel<0><<<dim3((unsigned)((rows / 16 + 3) / 4), (cd / 16 + 7) / 8),
                        dim3(256), 0, s>>>(
      sc.a, sc.w2h, prm(5), prm(6), prm(7), (void*)sc.hmlp, (int)rows, cd, cd);
  tmat_kernel<K><<<dim3(BP), dim3(((K * K + 31) / 32) * 32), 0, s>>>(
      sc.rel, prm(8), prm(9), prm(10), prm(11), prm(12), prm(13), prm(14), prm(15),
      prm(16), prm(17), prm(18), prm(19), sc.T);
  xtdw_kernel<K><<<dim3(BP), dim3(256), 0, s>>>(
      sc.hmlp, xprev, sc.idx, sc.T, prm(20), prm(21), sc.feat, Bn, P, cd, cin, dm, Kpad);
  convwT_kernel<<<dim3((unsigned)(((long)cout * Kpad + 255) / 256)), dim3(256), 0, s>>>(
      prm(22), sc.linw, Cdm, Kpad, cout);
  gemm_wmma_kernel<1><<<dim3((BP / 16 + 3) / 4, (cout / 16 + 7) / 8), dim3(256), 0, s>>>(
      sc.feat, sc.linw, prm(23), nullptr, nullptr, (void*)hout, BP, cout, Kpad);
}

extern "C" void kernel_launch(void* const* d_in, const int* in_sizes, int n_in,
                              void* d_out, int out_size, void* d_ws, size_t ws_size,
                              hipStream_t stream) {
  (void)in_sizes; (void)n_in; (void)out_size; (void)ws_size;
  const float* pos = (const float*)d_in[0];
  const float* x0  = (const float*)d_in[1];
  const int B = 16;

  size_t off = 0;
  auto alloc = [&](size_t bytes) -> void* {
    void* p = (char*)d_ws + off;
    off += (bytes + 255) & ~(size_t)255;
    return p;
  };
  float* h1   = (float*)alloc((size_t)B * 2048 * 128 * 4);
  float* pos2 = (float*)alloc((size_t)B * 768 * 3 * 4);
  float* h1s  = (float*)alloc((size_t)B * 768 * 128 * 4);
  float* h2   = (float*)alloc((size_t)B * 768 * 256 * 4);
  float* pos3 = (float*)alloc((size_t)B * 257 * 3 * 4);
  float* h2s  = (float*)alloc((size_t)B * 257 * 256 * 4);
  float* h3   = (float*)alloc((size_t)B * 257 * 512 * 4);
  float* h4   = (float*)alloc((size_t)B * 257 * 1024 * 4);
  int* fpi1 = (int*)alloc((size_t)B * 768 * 4);
  int* fpi2 = (int*)alloc((size_t)B * 257 * 4);

  Scratch sc;
  sc.idx  = (int*)alloc((size_t)B * 2048 * 8 * 4);        // max B*P*K
  sc.rel  = (float*)alloc((size_t)B * 2048 * 8 * 3 * 4);  // max B*P*K*3
  sc.a    = (_Float16*)alloc((size_t)16842752 * 2);       // max rows*cd (c4)
  sc.hmlp = (_Float16*)alloc((size_t)16842752 * 2);
  sc.T    = (float*)alloc((size_t)2097152 * 4);           // max B*P*K*K (c1)
  sc.feat = (_Float16*)alloc((size_t)6316032 * 2);        // max B*P*Kpad (c4)
  sc.w2h  = (_Float16*)alloc((size_t)256 * 256 * 2);
  sc.linw = (_Float16*)alloc((size_t)1536 * 1024 * 2);

  // ---- layer c1: P=2048, K=8, dil=1, cin=3, cd=32, dm=4, cout=128 ----
  run_layer<8, 1>(stream, d_in, 0, pos, 2048, x0, 3, 32, 4, 128, sc, h1);

  // FPS 2048 -> 768, gather pos + features
  fps_kernel<<<dim3(B), dim3(256), 0, stream>>>(pos, 2048, 768, fpi1);
  gather_kernel<<<dim3((unsigned)(((long)B * 768 * 3 + 255) / 256)), dim3(256), 0, stream>>>(
      pos, fpi1, pos2, B, 768, 3, 2048);
  gather_kernel<<<dim3((unsigned)(((long)B * 768 * 128 + 255) / 256)), dim3(256), 0, stream>>>(
      h1, fpi1, h1s, B, 768, 128, 2048);

  // ---- layer c2: P=768, K=12, dil=2, cin=128, cd=64, dm=2, cout=256 ----
  run_layer<12, 2>(stream, d_in, 1, pos2, 768, h1s, 128, 64, 2, 256, sc, h2);

  // FPS 768 -> 257
  fps_kernel<<<dim3(B), dim3(256), 0, stream>>>(pos2, 768, 257, fpi2);
  gather_kernel<<<dim3((unsigned)(((long)B * 257 * 3 + 255) / 256)), dim3(256), 0, stream>>>(
      pos2, fpi2, pos3, B, 257, 3, 768);
  gather_kernel<<<dim3((unsigned)(((long)B * 257 * 256 + 255) / 256)), dim3(256), 0, stream>>>(
      h2, fpi2, h2s, B, 257, 256, 768);

  // ---- layer c3: P=257, K=16, dil=2, cin=256, cd=128, dm=2, cout=512 ----
  run_layer<16, 2>(stream, d_in, 2, pos3, 257, h2s, 256, 128, 2, 512, sc, h3);

  // ---- layer c4: P=257, K=16, dil=2, cin=512, cd=256, dm=2, cout=1024 ----
  run_layer<16, 2>(stream, d_in, 3, pos3, 257, h3, 512, 256, 2, 1024, sc, h4);

  // global mean pool -> (16, 1024)
  meanpool_kernel<<<dim3((B * 1024 + 255) / 256), dim3(256), 0, stream>>>(
      h4, (float*)d_out, B, 257, 1024);
}
